// GraphTransformerNet_63565515981157
// MI455X (gfx1250) — matrix operand order repaired and checked by reading, not compiled, not run on hardware
//
#include <hip/hip_runtime.h>
#include <hip/hip_bf16.h>

typedef __attribute__((ext_vector_type(16))) _Float16 v16h;
typedef __attribute__((ext_vector_type(8)))  _Float16 v8h;
typedef __attribute__((ext_vector_type(8)))  float    v8f;

#define C_DIM   64
#define HEADS   4
#define HEAD_D  16
#define EDGE_D  16
#define NLAYERS 3
#define NGRAPHS 128

// ---------- ordered-float <-> uint mapping (for atomicMax on floats) ----------
__device__ __forceinline__ unsigned fmap_ord(float f) {
  unsigned u = __float_as_uint(f);
  return u ^ ((unsigned)((int)u >> 31) | 0x80000000u);
}
__device__ __forceinline__ float funmap_ord(unsigned m) {
  unsigned u = (m & 0x80000000u) ? (m ^ 0x80000000u) : ~m;
  return __uint_as_float(u);
}

// ---------- convert x -> h (f32) and h_half (f16) ----------
__global__ void k_convert(const float* __restrict__ x, float* __restrict__ h,
                          _Float16* __restrict__ hh, int n) {
  int i = blockIdx.x * blockDim.x + threadIdx.x;
  if (i < n) { float v = x[i]; h[i] = v; hh[i] = (_Float16)v; }
}

// ---------- pack a 64x64 f32 weight (row-major, K x N) into WMMA-B f16 layout ----------
// out[((nt*2+kt)*32 + lane)*16 + idx] = W[(kt*32 + (idx>=8?16:0) + (lane>=16?8:0) + (idx&7))*64 + nt*16 + (lane&15)]
__global__ void k_packB(const float* __restrict__ W, _Float16* __restrict__ out) {
  for (int t = threadIdx.x; t < 4096; t += blockDim.x) {
    int idx  = t & 15;
    int lane = (t >> 4) & 31;
    int kt   = (t >> 9) & 1;
    int nt   = t >> 10;
    int k    = kt * 32 + ((idx >> 3) * 16) + ((lane >> 4) * 8) + (idx & 7);
    int col  = nt * 16 + (lane & 15);
    out[t] = (_Float16)W[k * 64 + col];
  }
}

// ---------- fused 4-way node GEMM: q, k, v, and skip(->agg) via WMMA ----------
__global__ __launch_bounds__(256)
void k_gemm4(const _Float16* __restrict__ hh,
             const _Float16* __restrict__ packed,  // 4 matrices * 4096 halfs
             const float* __restrict__ bq, const float* __restrict__ bk,
             const float* __restrict__ bv, const float* __restrict__ bs,
             float* __restrict__ outq, float* __restrict__ outk,
             float* __restrict__ outv, float* __restrict__ outskip,
             int nNodes) {
  int gw   = (blockIdx.x * blockDim.x + threadIdx.x) >> 5;
  int lane = threadIdx.x & 31;
  int mtiles = nNodes >> 4;
  if (gw >= mtiles) return;
  int m0  = gw << 4;
  int row = m0 + (lane & 15);
  int koff = (lane >> 4) << 3;                 // 0 or 8 halfs
  const _Float16* ap = hh + (size_t)row * 64;

  // A in 16-bit 16x32 WMMA layout: lane holds two contiguous 8-half chunks
  union U { v16h v; struct { v8h lo, hi; } s; };
  U A0, A1;
  A0.s.lo = *(const v8h*)(ap + koff);
  A0.s.hi = *(const v8h*)(ap + koff + 16);
  A1.s.lo = *(const v8h*)(ap + 32 + koff);
  A1.s.hi = *(const v8h*)(ap + 48 + koff);

  float* outs[4]         = {outq, outk, outv, outskip};
  const float* biases[4] = {bq, bk, bv, bs};
  int colL  = lane & 15;
  int rbase = m0 + ((lane >> 4) << 3);

  #pragma unroll
  for (int mat = 0; mat < 4; ++mat) {
    const _Float16* pw = packed + mat * 4096;
    float* op = outs[mat];
    const float* bp = biases[mat];
    #pragma unroll
    for (int nt = 0; nt < 4; ++nt) {
      v16h B0 = *(const v16h*)(pw + ((nt * 2 + 0) * 32 + lane) * 16);
      v16h B1 = *(const v16h*)(pw + ((nt * 2 + 1) * 32 + lane) * 16);
      v8f acc = {};
      acc = __builtin_amdgcn_wmma_f32_16x16x32_f16(false, A0.v, false, B0,
                                                   (short)0, acc, false, false);
      acc = __builtin_amdgcn_wmma_f32_16x16x32_f16(false, A1.v, false, B1,
                                                   (short)0, acc, false, false);
      int col = nt * 16 + colL;
      float bb = bp[col];
      #pragma unroll
      for (int r = 0; r < 8; ++r)
        op[(size_t)(rbase + r) * 64 + col] = acc[r] + bb;
    }
  }
}

// ---------- shared helper: load We/be to LDS ----------
__device__ __forceinline__ void load_We_lds(float* sWe, const float* We, const float* be) {
  for (int t = threadIdx.x; t < EDGE_D * 64 + 64; t += blockDim.x)
    sWe[t] = (t < EDGE_D * 64) ? We[t] : be[t - EDGE_D * 64];
  __syncthreads();
}

// ---------- pass 1: per-edge logits + per-(dst,head) running max ----------
__global__ __launch_bounds__(256)
void k_edge_logits(const float* __restrict__ q, const float* __restrict__ kbuf,
                   const int* __restrict__ ei, const float* __restrict__ edge_attr,
                   const float* __restrict__ We, const float* __restrict__ be,
                   float* __restrict__ logits, unsigned* __restrict__ m_u,
                   int nEdges) {
  __shared__ float sWe[EDGE_D * 64 + 64];
  load_We_lds(sWe, We, be);
  int e    = (blockIdx.x * blockDim.x + threadIdx.x) >> 5;
  int lane = threadIdx.x & 31;
  if (e >= nEdges) return;
  int src = ei[e];
  int dst = ei[nEdges + e];
  int c0  = lane * 2;
  float2 qv = *(const float2*)(q + (size_t)dst * 64 + c0);
  float2 kv = *(const float2*)(kbuf + (size_t)src * 64 + c0);
  float aL = (lane < EDGE_D) ? edge_attr[(size_t)e * EDGE_D + lane] : 0.0f;
  float ea0 = sWe[EDGE_D * 64 + c0];
  float ea1 = sWe[EDGE_D * 64 + c0 + 1];
  #pragma unroll
  for (int d = 0; d < EDGE_D; ++d) {
    float ad = __shfl(aL, d);
    ea0 += ad * sWe[d * 64 + c0];
    ea1 += ad * sWe[d * 64 + c0 + 1];
  }
  float part = qv.x * (kv.x + ea0) + qv.y * (kv.y + ea1);
  part += __shfl_xor(part, 1);
  part += __shfl_xor(part, 2);
  part += __shfl_xor(part, 4);
  if ((lane & 7) == 0) {
    int h = lane >> 3;
    float lg = part * 0.25f;                 // 1/sqrt(HEAD_D)
    logits[(size_t)e * 4 + h] = lg;
    atomicMax(&m_u[(size_t)dst * 4 + h], fmap_ord(lg));
  }
}

// ---------- pass 2: exp(logit - max) and denominator accumulation ----------
__global__ __launch_bounds__(256)
void k_edge_exp(const int* __restrict__ ei, float* __restrict__ logits,
                const unsigned* __restrict__ m_u, float* __restrict__ denom,
                int nEdges) {
  int t = blockIdx.x * blockDim.x + threadIdx.x;
  if (t >= nEdges * 4) return;
  int e = t >> 2, h = t & 3;
  int dst = ei[nEdges + e];
  float ex = __expf(logits[t] - funmap_ord(m_u[(size_t)dst * 4 + h]));
  logits[t] = ex;
  atomicAdd(&denom[(size_t)dst * 4 + h], ex);
}

// ---------- pass 3: scatter alpha*(v+ea) into agg (which holds skip) ----------
__global__ __launch_bounds__(256)
void k_edge_agg(const float* __restrict__ vbuf, const int* __restrict__ ei,
                const float* __restrict__ edge_attr,
                const float* __restrict__ We, const float* __restrict__ be,
                const float* __restrict__ logits, const float* __restrict__ denom,
                float* __restrict__ agg, int nEdges) {
  __shared__ float sWe[EDGE_D * 64 + 64];
  load_We_lds(sWe, We, be);
  int e    = (blockIdx.x * blockDim.x + threadIdx.x) >> 5;
  int lane = threadIdx.x & 31;
  if (e >= nEdges) return;
  int src = ei[e];
  int dst = ei[nEdges + e];
  float exv = 0.0f, dnv = 1.0f;
  if (lane < 4) {
    exv = logits[(size_t)e * 4 + lane];
    dnv = denom[(size_t)dst * 4 + lane];
  }
  float exb = __shfl(exv, lane >> 3);
  float dnb = __shfl(dnv, lane >> 3);
  float alpha = exb / fmaxf(dnb, 1e-16f);
  int c0 = lane * 2;
  float2 vv = *(const float2*)(vbuf + (size_t)src * 64 + c0);
  float aL = (lane < EDGE_D) ? edge_attr[(size_t)e * EDGE_D + lane] : 0.0f;
  float ea0 = sWe[EDGE_D * 64 + c0];
  float ea1 = sWe[EDGE_D * 64 + c0 + 1];
  #pragma unroll
  for (int d = 0; d < EDGE_D; ++d) {
    float ad = __shfl(aL, d);
    ea0 += ad * sWe[d * 64 + c0];
    ea1 += ad * sWe[d * 64 + c0 + 1];
  }
  atomicAdd(&agg[(size_t)dst * 64 + c0],     alpha * (vv.x + ea0));
  atomicAdd(&agg[(size_t)dst * 64 + c0 + 1], alpha * (vv.y + ea1));
}

// ---------- layer epilogue: optional ReLU, refresh f32 + f16 node features ----------
__global__ void k_update(const float* __restrict__ agg, float* __restrict__ h,
                         _Float16* __restrict__ hh, int n, int doRelu) {
  int i = blockIdx.x * blockDim.x + threadIdx.x;
  if (i < n) {
    float v = agg[i];
    if (doRelu) v = fmaxf(v, 0.0f);
    h[i] = v;
    hh[i] = (_Float16)v;
  }
}

// ---------- global mean pool (accumulate) ----------
__global__ __launch_bounds__(256)
void k_pool(const float* __restrict__ h, const int* __restrict__ batch,
            float* __restrict__ sums, float* __restrict__ cnt, int nNodes) {
  int t = blockIdx.x * blockDim.x + threadIdx.x;
  if (t >= nNodes * 32) return;
  int n = t >> 5, l = t & 31;
  int g = batch[n];
  float2 v = *(const float2*)(h + (size_t)n * 64 + l * 2);
  atomicAdd(&sums[(size_t)g * 64 + 2 * l],     v.x);
  atomicAdd(&sums[(size_t)g * 64 + 2 * l + 1], v.y);
  if (l == 0) atomicAdd(&cnt[g], 1.0f);
}

// ---------- readout MLP: relu(pooled@W1+b1)@W2+b2 ----------
__global__ void k_head(const float* __restrict__ sums, const float* __restrict__ cnt,
                       const float* __restrict__ w1, const float* __restrict__ b1,
                       const float* __restrict__ w2, const float* __restrict__ b2,
                       float* __restrict__ out) {
  int g = blockIdx.x, c = threadIdx.x;  // 64 threads
  __shared__ float pl[64];
  __shared__ float red[64];
  pl[c] = sums[(size_t)g * 64 + c] / fmaxf(cnt[g], 1.0f);
  __syncthreads();
  float acc = b1[c];
  #pragma unroll
  for (int k = 0; k < 64; ++k) acc += pl[k] * w1[k * 64 + c];
  red[c] = fmaxf(acc, 0.0f) * w2[c];
  __syncthreads();
  if (c == 0) {
    float s = b2[0];
    for (int k = 0; k < 64; ++k) s += red[k];
    out[g] = s;
  }
}

extern "C" void kernel_launch(void* const* d_in, const int* in_sizes, int n_in,
                              void* d_out, int out_size, void* d_ws, size_t ws_size,
                              hipStream_t stream) {
  const float* x         = (const float*)d_in[0];
  const int*   ei        = (const int*)d_in[1];
  const float* edge_attr = (const float*)d_in[2];
  const int*   batch     = (const int*)d_in[3];
  const float* Wq = (const float*)d_in[4];   const float* bq = (const float*)d_in[5];
  const float* Wk = (const float*)d_in[6];   const float* bk = (const float*)d_in[7];
  const float* Wv = (const float*)d_in[8];   const float* bv = (const float*)d_in[9];
  const float* We = (const float*)d_in[10];  const float* be = (const float*)d_in[11];
  const float* Ws = (const float*)d_in[12];  const float* bs = (const float*)d_in[13];
  const float* l1w = (const float*)d_in[14]; const float* l1b = (const float*)d_in[15];
  const float* l2w = (const float*)d_in[16]; const float* l2b = (const float*)d_in[17];

  int nNodes = in_sizes[0] / C_DIM;
  int nEdges = in_sizes[1] / 2;
  int ntot   = nNodes * C_DIM;

  // workspace carve (256B aligned)
  char* p = (char*)d_ws;
  auto alloc = [&](size_t bytes) -> char* {
    char* r = p; p += (bytes + 255) & ~(size_t)255; return r;
  };
  size_t NB = (size_t)ntot * sizeof(float);
  float*    q      = (float*)alloc(NB);
  float*    kb     = (float*)alloc(NB);
  float*    vb     = (float*)alloc(NB);
  float*    agg    = (float*)alloc(NB);
  float*    h      = (float*)alloc(NB);
  _Float16* hh     = (_Float16*)alloc((size_t)ntot * 2);
  float*    logits = (float*)alloc((size_t)nEdges * 4 * sizeof(float));
  unsigned* m_u    = (unsigned*)alloc((size_t)nNodes * 4 * sizeof(unsigned));
  float*    denom  = (float*)alloc((size_t)nNodes * 4 * sizeof(float));
  _Float16* packed = (_Float16*)alloc((size_t)NLAYERS * 4 * 4096 * 2);
  float*    sums   = (float*)alloc((size_t)NGRAPHS * 64 * sizeof(float));
  float*    cnt    = (float*)alloc((size_t)NGRAPHS * sizeof(float));

  // pre-swizzle all layer weights into WMMA-B layout (f16)
  for (int l = 0; l < NLAYERS; ++l) {
    k_packB<<<1, 256, 0, stream>>>(Wq + l * 4096, packed + (l * 4 + 0) * 4096);
    k_packB<<<1, 256, 0, stream>>>(Wk + l * 4096, packed + (l * 4 + 1) * 4096);
    k_packB<<<1, 256, 0, stream>>>(Wv + l * 4096, packed + (l * 4 + 2) * 4096);
    k_packB<<<1, 256, 0, stream>>>(Ws + l * 4096, packed + (l * 4 + 3) * 4096);
  }

  k_convert<<<(ntot + 255) / 256, 256, 0, stream>>>(x, h, hh, ntot);

  int mtiles     = nNodes / 16;
  int gemmBlocks = (mtiles * 32 + 255) / 256;
  int edgeBlocks = (int)(((size_t)nEdges * 32 + 255) / 256);

  for (int l = 0; l < NLAYERS; ++l) {
    hipMemsetAsync(m_u,   0, (size_t)nNodes * 4 * sizeof(unsigned), stream);
    hipMemsetAsync(denom, 0, (size_t)nNodes * 4 * sizeof(float),    stream);
    k_gemm4<<<gemmBlocks, 256, 0, stream>>>(hh, packed + (size_t)l * 4 * 4096,
        bq + l * 64, bk + l * 64, bv + l * 64, bs + l * 64,
        q, kb, vb, agg, nNodes);
    k_edge_logits<<<edgeBlocks, 256, 0, stream>>>(q, kb, ei, edge_attr,
        We + l * EDGE_D * 64, be + l * 64, logits, m_u, nEdges);
    k_edge_exp<<<(nEdges * 4 + 255) / 256, 256, 0, stream>>>(ei, logits, m_u,
        denom, nEdges);
    k_edge_agg<<<edgeBlocks, 256, 0, stream>>>(vb, ei, edge_attr,
        We + l * EDGE_D * 64, be + l * 64, logits, denom, agg, nEdges);
    k_update<<<(ntot + 255) / 256, 256, 0, stream>>>(agg, h, hh, ntot, l < NLAYERS - 1);
  }

  hipMemsetAsync(sums, 0, (size_t)NGRAPHS * 64 * sizeof(float), stream);
  hipMemsetAsync(cnt,  0, (size_t)NGRAPHS * sizeof(float),      stream);
  k_pool<<<(int)(((size_t)nNodes * 32 + 255) / 256), 256, 0, stream>>>(h, batch,
      sums, cnt, nNodes);
  k_head<<<NGRAPHS, 64, 0, stream>>>(sums, cnt, l1w, l1b, l2w, l2b, (float*)d_out);
}
